// LIAFConvCell_34600256537576
// MI455X (gfx1250) — matrix operand order repaired and checked by reading, not compiled, not run on hardware
//
#include <hip/hip_runtime.h>
#include <hip/hip_bf16.h>

typedef __attribute__((ext_vector_type(16))) _Float16 v16h;
typedef __attribute__((ext_vector_type(2)))  __fp16   fp16x2;  // cvt_pkrtz result type
typedef __attribute__((ext_vector_type(8)))  float    v8f;

// x tile in LDS: f32, [C=2][18 rows][stride 12], double buffered.
// Columns 10..11 of each row are never staged -> stay zero (zero slot lives there).
#define XS    12
#define XBUF  (2 * 18 * XS)   // 432 floats per buffer
#define ZIDX  10              // row 0, col 10: guaranteed-zero cell (buffer-relative)

// Issue async global->LDS copies of timestep TT's x tile into buffer BUFSEL.
// Per-lane predication via divergent if (EXEC-masked); tracked by ASYNCcnt.
#define ISSUE_ASYNC(TT, BUFSEL)                                               \
    {                                                                         \
        _Pragma("unroll")                                                     \
        for (int j = 0; j < 2; ++j) {                                         \
            if (vld[j]) {                                                     \
                unsigned la = lbyte[j] + (BUFSEL) * (XBUF * 4);               \
                unsigned long long ga =                                       \
                    gaddr0[j] + ((unsigned long long)(TT) << 16);             \
                asm volatile("global_load_async_to_lds_b32 %0, %1, off"       \
                             :: "v"(la), "v"(ga) : "memory");                 \
            }                                                                 \
        }                                                                     \
    }

__global__ __launch_bounds__(256, 1) void liaf_fused_wmma_kernel(
    const float* __restrict__ x,      // (16,2,16,128,128)
    const float* __restrict__ Wk,     // (64,2,3,3)
    const float* __restrict__ bias,   // (64,)
    const float* __restrict__ decay,  // (1,)
    float*       __restrict__ out)    // (16,64,16,64,64)
{
    __shared__ _Float16 wlds[32 * 64];   // [k padded to 32][o=64]
    __shared__ float    xbuf[2 * XBUF];  // double-buffered x tile

    const int tid   = threadIdx.x;
    const int lane  = tid & 31;
    const int wid   = tid >> 5;
    const int waveY = wid & 3;   // 4 wave-rows -> 16 pixel rows
    const int waveX = wid >> 2;  // 2 wave-cols -> 8 pixel cols

    int blk = blockIdx.x;
    const int tw = blk & 15; blk >>= 4;   // 16 w-tiles of 8
    const int th = blk & 7;  blk >>= 3;   // 8  h-tiles of 16
    const int b  = blk;                   // 16 batches
    const int gy0 = th * 16;
    const int gx0 = tw * 8;

    // ---- zero both buffers once (covers halo OOB cells + padding cols) ----
    for (int i = tid; i < 2 * XBUF; i += 256) xbuf[i] = 0.0f;

    // ---- weights -> LDS (K-major, f16, zero-padded K=18..31) ----
    for (int i = tid; i < 32 * 64; i += 256) {
        const int k = i >> 6, o = i & 63;
        wlds[i] = (_Float16)((k < 18) ? Wk[o * 18 + k] : 0.0f);
    }

    const float dnorm = 1.0f / (1.0f + expf(-decay[0]));   // sigmoid(decay)

    // ---- t-invariant staging descriptors: 2 halo elements per thread ----
    unsigned           lbyte[2];
    unsigned long long gaddr0[2];
    bool               vld[2];
    #pragma unroll
    for (int j = 0; j < 2; ++j) {
        const int i = tid + 256 * j;
        const bool act = (i < 2 * 18 * 10);
        int c = 0, yy = 0, xx = 0;
        if (act) {
            c = i / 180;
            const int r = i - c * 180;
            yy = r / 10;
            xx = r - yy * 10;
        }
        const int gy = gy0 - 1 + yy;
        const int gx = gx0 - 1 + xx;
        vld[j] = act && gy >= 0 && gy < 128 && gx >= 0 && gx < 128;
        // LDS byte address (buffer 0); flat low 32 bits == LDS offset
        lbyte[j] = (unsigned)(uintptr_t)&xbuf[(c * 18 + yy) * XS + xx];
        const int gyc = gy < 0 ? 0 : gy, gxc = gx < 0 ? 0 : gx;
        gaddr0[j] = (unsigned long long)(uintptr_t)
                    (x + ((((size_t)b * 2 + c) * 16) << 14)
                       + ((size_t)gyc << 7) + gxc);       // t=0; per t add t<<16 B
    }

    // ---- per-lane im2col gather offsets (t-invariant, buffer-relative) ----
    const int M  = lane & 15;            // A-matrix pixel row
    const int kb = (lane >> 4) << 3;     // K base (16-bit A VGPR layout)
    const int py = waveY * 4 + (M >> 2);
    const int px = waveX * 4 + (M & 3);

    int aoff[16];
    #pragma unroll
    for (int e = 0; e < 16; ++e) {
        const int K = kb + e + ((e >= 8) ? 8 : 0);
        if (K < 18) {
            const int c  = (K >= 9) ? 1 : 0;
            const int r  = K - c * 9;
            const int kh = r / 3;
            const int kw = r - kh * 3;
            aoff[e] = (c * 18 + (py + kh)) * XS + (px + kw);
        } else {
            aoff[e] = ZIDX;              // always-zero padding cell
        }
    }

    const int PH = ((gy0 + waveY * 4) >> 1) + (lane >> 4);  // pooled row
    const int PW = (gx0 + waveX * 4) >> 1;                  // pooled col (even)

    __syncthreads();   // zeros + weights visible to everyone

    // kick off async staging of t=0 into buffer 0 (overlaps with B-frag build)
    ISSUE_ASYNC(0, 0);

    // ---- B fragments: lane%16 = output channel, 16 f16 = K (table order) ----
    v16h bfrag[4];
    {
        const _Float16* wrow = &wlds[(lane & 15) + ((lane >> 4) << 9)];
        #pragma unroll
        for (int g = 0; g < 4; ++g) {
            #pragma unroll
            for (int e = 0; e < 16; ++e) {
                bfrag[g][e] = wrow[(e + ((e >= 8) ? 8 : 0)) * 64 + g * 16];
            }
        }
    }

    float blane[4];
    #pragma unroll
    for (int g = 0; g < 4; ++g) blane[g] = bias[(lane & 15) + g * 16];

    // ---- membrane state (spike-masked) lives in VGPRs across T ----
    v8f memk[4];
    #pragma unroll
    for (int g = 0; g < 4; ++g) {
        #pragma unroll
        for (int r = 0; r < 8; ++r) memk[g][r] = 0.0f;
    }

    #pragma unroll 2   // buffer parity becomes compile-time
    for (int t = 0; t < 16; ++t) {
        const int rb = t & 1;

        asm volatile("s_wait_asynccnt 0x0" ::: "memory"); // this wave's copies done
        __syncthreads();                                   // all waves' copies visible

        if (t + 1 < 16) ISSUE_ASYNC(t + 1, (t + 1) & 1);   // overlap next fetch

        // ---- gather 16x32 f16 A fragment: f32 LDS reads + pair-pack ----
        const float* xb = &xbuf[rb * XBUF];
        float av[16];
        #pragma unroll
        for (int e = 0; e < 16; ++e) av[e] = xb[aoff[e]];

        union { v16h v; fp16x2 p[8]; } au;
        #pragma unroll
        for (int q = 0; q < 8; ++q) {
            au.p[q] = __builtin_amdgcn_cvt_pkrtz(av[2 * q], av[2 * q + 1]);
        }

        // ---- 4 WMMAs cover 64 channels; C operand carries the recurrence ----
        #pragma unroll
        for (int g = 0; g < 4; ++g) {
            v8f cfr;
            #pragma unroll
            for (int r = 0; r < 8; ++r) {
                // mem_prev * d * (1 - spike_prev) + bias (mask folded in memk)
                cfr[r] = fmaf(memk[g][r], dnorm, blane[g]);
            }

            // D = conv(x_t) + bias + decayed-masked-membrane
            v8f dfr = __builtin_amdgcn_wmma_f32_16x16x32_f16(
                false, au.v, false, bfrag[g], (short)0, cfr, false, false);

            float s[8];
            #pragma unroll
            for (int r = 0; r < 8; ++r) {
                const bool fire = dfr[r] > 0.5f;       // LIF threshold
                s[r]       = fire ? 0.25f : 0.0f;      // spike * 0.25 (pool fold)
                memk[g][r] = fire ? 0.0f : dfr[r];     // mem * (1 - spike)
            }

            // 2x2 avg-pool: rows {0,1,4,5} and {2,3,6,7} of the D tile
            const float p0 = s[0] + s[1] + s[4] + s[5];
            const float p1 = s[2] + s[3] + s[6] + s[7];

            const int o = (lane & 15) + g * 16;
            const size_t oi = ((((size_t)b * 64 + o) * 16 + t) << 12)
                            + ((size_t)PH << 6) + PW;
            float2 pv; pv.x = p0; pv.y = p1;
            *(float2*)(out + oi) = pv;   // 8B-aligned (PW even)
        }
    }
}

extern "C" void kernel_launch(void* const* d_in, const int* in_sizes, int n_in,
                              void* d_out, int out_size, void* d_ws, size_t ws_size,
                              hipStream_t stream) {
    const float* x     = (const float*)d_in[0];  // (16,2,16,128,128)
    const float* Wk    = (const float*)d_in[1];  // (64,2,3,3)
    const float* bias  = (const float*)d_in[2];  // (64,)
    const float* decay = (const float*)d_in[3];  // (1,)
    float* out = (float*)d_out;                  // (16,64,16,64,64)

    // 16 batches * 8 h-tiles(16) * 16 w-tiles(8) = 2048 WGs, 8 wave32 each
    liaf_fused_wmma_kernel<<<dim3(16 * 8 * 16), dim3(256), 0, stream>>>(
        x, Wk, bias, decay, out);
}
